// GraphMSG_43593918055121
// MI455X (gfx1250) — matrix operand order repaired and checked by reading, not compiled, not run on hardware
//
#include <hip/hip_runtime.h>
#include <hip/hip_bf16.h>

// ---------------------------------------------------------------------------
// GraphMSG (encoder -> processor -> decoder GNN) for gfx1250 / MI455X.
// All dense MLPs run as fused 2-layer WMMA GEMMs (f16 in, f32 accumulate)
// using the codegen-verified __builtin_amdgcn_wmma_f32_16x16x32_f16.
// Gather-concat edge inputs are fused into the GEMM A-tile load; scatter-sum
// uses global f32 atomics. Scratch is bump-allocated from d_ws.
// ---------------------------------------------------------------------------

typedef _Float16 half_t;
typedef __attribute__((ext_vector_type(16))) _Float16 v16h;
typedef __attribute__((ext_vector_type(8)))  _Float16 v8h;
typedef __attribute__((ext_vector_type(8)))  float    v8f;

#define THREADS 256
#define WAVES   8
#define ROWS    16
#define KPAD_MAX   416   // max padded K (proc edge mlp: 396 -> 416)
#define M1PAD_MAX  160   // max padded hidden dim (132 -> 160)
#define M2_MAX     144

__device__ __forceinline__ float silu_f(float x) {
  return x * (1.0f / (1.0f + __expf(-x)));
}

// ---------------------------------------------------------------------------
// Fused 2-layer MLP:  out = [LN]( act2( act1(X @ W0 + b0) @ W1 + b1 ) ) + resid
// X row is composed of up to 3 segments, each optionally gathered via an
// index array (edge gather / node concat are free).
// W0t: f16 [rup16(M1), Kpad]   (transposed, zero padded)
// W1t: f16 [rup16(M2), M1pad]  (transposed, zero padded)
// One block = 16 rows; 8 waves each own 16-column output slabs.
// ---------------------------------------------------------------------------
__global__ void __launch_bounds__(THREADS)
mlp2_wmma(const float* __restrict__ s0p, const int* __restrict__ s0i, int sw0,
          const float* __restrict__ s1p, const int* __restrict__ s1i, int sw1,
          const float* __restrict__ s2p, const int* __restrict__ s2i, int sw2,
          int N, int K, int Kpad,
          const half_t* __restrict__ W0t, const float* __restrict__ b0v,
          int M1, int M1pad,
          const half_t* __restrict__ W1t, const float* __restrict__ b1v,
          int M2,
          const float* __restrict__ lng, const float* __restrict__ lnb,
          const float* __restrict__ resid, int resid_stride,
          float* __restrict__ out, int final_act)
{
  __shared__ __align__(16) half_t Xs[ROWS * KPAD_MAX];
  __shared__ __align__(16) half_t Hs[ROWS * M1PAD_MAX];
  __shared__ float Os[ROWS * M2_MAX];
  __shared__ float s_mu[ROWS], s_rs[ROWS];

  const int tid  = threadIdx.x;
  const int wave = tid >> 5;
  const int lane = tid & 31;
  const int row0 = blockIdx.x * ROWS;

  // Zero hidden staging (padded K region of layer 2 must be clean f16 zeros).
  for (int i = tid; i < ROWS * M1pad; i += THREADS) Hs[i] = (half_t)0.0f;

  // Stage X tile (16 x Kpad) into LDS as f16, fusing segment concat + gather.
  for (int i = tid; i < ROWS * Kpad; i += THREADS) {
    int r = i / Kpad, k = i - r * Kpad;
    int gr = row0 + r;
    float v = 0.0f;
    if (gr < N && k < K) {
      if (k < sw0) {
        int g = s0i ? s0i[gr] : gr;
        v = s0p[(long long)g * sw0 + k];
      } else if (k < sw0 + sw1) {
        int g = s1i ? s1i[gr] : gr;
        v = s1p[(long long)g * sw1 + (k - sw0)];
      } else {
        int g = s2i ? s2i[gr] : gr;
        v = s2p[(long long)g * sw2 + (k - sw0 - sw1)];
      }
    }
    Xs[i] = (half_t)v;
  }
  __syncthreads();

  const int arow = lane & 15;     // A-fragment row / B-fragment column offset
  const int ahi  = lane >> 4;     // half-wave selector

  // ---- layer 1: H = silu(X @ W0 + b0) -> LDS f16 ----
  for (int mt = wave; mt * 16 < M1; mt += WAVES) {
    v8f acc = {0.f, 0.f, 0.f, 0.f, 0.f, 0.f, 0.f, 0.f};
    const int col = mt * 16 + arow;
    for (int k0 = 0; k0 < Kpad; k0 += 32) {
      // k order used here for A and B (not the ISA table order): lanes 0-15 K=[k0..k0+15],
      //                     lanes 16-31 K=[k0+16..k0+31]
      v8h alo = *(const v8h*)&Xs[arow * Kpad + k0 + ahi * 16];
      v8h ahiV = *(const v8h*)&Xs[arow * Kpad + k0 + ahi * 16 + 8];
      v16h a = __builtin_shufflevector(alo, ahiV,
                 0,1,2,3,4,5,6,7,8,9,10,11,12,13,14,15);
      // B 32x16: lane holds 16 consecutive K for its column -> contiguous W^T
      v16h b = *(const v16h*)&W0t[(long long)col * Kpad + k0 + ahi * 16];
      acc = __builtin_amdgcn_wmma_f32_16x16x32_f16(
                false, a, false, b, (short)0, acc, false, false);
    }
    if (col < M1) {
      float bb = b0v[col];
      for (int r = 0; r < 8; ++r) {
        int rr = r + ahi * 8;          // D: lane 0-15 -> M=r, lane 16-31 -> M=r+8
        float y = silu_f(acc[r] + bb);
        Hs[rr * M1pad + col] = (half_t)y;
      }
    }
  }
  __syncthreads();

  // ---- layer 2: O = [silu](H @ W1 + b1) -> LDS f32 ----
  for (int mt = wave; mt * 16 < M2; mt += WAVES) {
    v8f acc = {0.f, 0.f, 0.f, 0.f, 0.f, 0.f, 0.f, 0.f};
    const int col = mt * 16 + arow;
    for (int k0 = 0; k0 < M1pad; k0 += 32) {
      v8h alo = *(const v8h*)&Hs[arow * M1pad + k0 + ahi * 16];
      v8h ahiV = *(const v8h*)&Hs[arow * M1pad + k0 + ahi * 16 + 8];
      v16h a = __builtin_shufflevector(alo, ahiV,
                 0,1,2,3,4,5,6,7,8,9,10,11,12,13,14,15);
      v16h b = *(const v16h*)&W1t[(long long)col * M1pad + k0 + ahi * 16];
      acc = __builtin_amdgcn_wmma_f32_16x16x32_f16(
                false, a, false, b, (short)0, acc, false, false);
    }
    if (col < M2) {
      float bb = b1v[col];
      for (int r = 0; r < 8; ++r) {
        int rr = r + ahi * 8;
        float y = acc[r] + bb;
        if (final_act) y = silu_f(y);
        Os[rr * M2 + col] = y;
      }
    }
  }
  __syncthreads();

  // ---- LayerNorm statistics (one thread per row) ----
  if (lng && tid < ROWS) {
    float s = 0.0f;
    for (int c = 0; c < M2; ++c) s += Os[tid * M2 + c];
    float mu = s / (float)M2;
    float v = 0.0f;
    for (int c = 0; c < M2; ++c) { float d = Os[tid * M2 + c] - mu; v += d * d; }
    s_mu[tid] = mu;
    s_rs[tid] = rsqrtf(v / (float)M2 + 1e-5f);
  }
  __syncthreads();

  // ---- epilogue: LN + residual + store ----
  for (int i = tid; i < ROWS * M2; i += THREADS) {
    int r = i / M2, c = i - r * M2;
    int gr = row0 + r;
    if (gr >= N) continue;
    float y = Os[i];
    if (lng) y = (y - s_mu[r]) * s_rs[r] * lng[c] + lnb[c];
    if (resid) y += resid[(long long)gr * resid_stride + c];
    out[(long long)gr * M2 + c] = y;
  }
}

// ---------------------------------------------------------------------------
// Helper kernels
// ---------------------------------------------------------------------------

// Transpose + f32->f16 + pad: W[K,M] (row major) -> Wt[Mpad, Kpad] f16.
__global__ void k_wconv(const float* __restrict__ W, int K, int M,
                        int Kpad, int Mpad, half_t* __restrict__ out) {
  int i = blockIdx.x * THREADS + threadIdx.x;
  if (i >= Mpad * Kpad) return;
  int m = i / Kpad, k = i - m * Kpad;
  out[i] = (m < M && k < K) ? (half_t)W[(long long)k * M + m] : (half_t)0.0f;
}

// x_era build: [n, 0:180]=x permuted (m,f), [180:184]=latlon, [184:192]=trainable
__global__ void k_build_xera(const float* __restrict__ x,
                             const float* __restrict__ ll,
                             const float* __restrict__ tr,
                             float* __restrict__ out, int N) {
  int i = blockIdx.x * THREADS + threadIdx.x;
  if (i >= N * 192) return;
  int n = i / 192, k = i - n * 192;
  float v;
  if (k < 180) { int m = k / 90, f = k - m * 90; v = x[((long long)m * 40320 + n) * 90 + f]; }
  else if (k < 184) v = ll[n * 4 + (k - 180)];
  else v = tr[n * 8 + (k - 184)];
  out[i] = v;
}

__global__ void k_concat2(const float* __restrict__ a, int wa,
                          const float* __restrict__ b, int wb,
                          float* __restrict__ out, int N) {
  int W = wa + wb;
  int i = blockIdx.x * THREADS + threadIdx.x;
  if (i >= N * W) return;
  int n = i / W, k = i - n * W;
  out[i] = (k < wa) ? a[(long long)n * wa + k] : b[(long long)n * wb + (k - wa)];
}

// x_comb = x_proc[:, :128] + 2 * x_latent   (double residual in source)
__global__ void k_combine(const float* __restrict__ xp,
                          const float* __restrict__ xl,
                          float* __restrict__ out, int N) {
  int i = blockIdx.x * THREADS + threadIdx.x;
  if (i >= N * 128) return;
  int n = i >> 7, c = i & 127;
  out[i] = xp[n * 132 + c] + 2.0f * xl[i];
}

// segment_sum via f32 atomics into a zeroed buffer
__global__ void k_scatter_add(const float* __restrict__ e,
                              const int* __restrict__ dst,
                              int E, int C, float* __restrict__ agg) {
  int i = blockIdx.x * THREADS + threadIdx.x;
  if (i >= E * C) return;
  int ed = i / C, c = i - ed * C;
  atomicAdd(&agg[(long long)dst[ed] * C + c], e[i]);
}

// ---------------------------------------------------------------------------
// Host side
// ---------------------------------------------------------------------------
static inline int rup(int x, int a) { return (x + a - 1) / a * a; }

struct MlpH {
  const half_t *w0t, *w1t;
  const float *b0, *b1, *g, *bn;
  int K, Kpad, M1, M1pad, M2;
};

// Input flattening (insertion order of setup_inputs() dict, params recursively
// in declaration order; each MLP = w0,w1,b0,b1[,ln.g,ln.b]):
//   0: x   1-5: era/h/e2h/h2h/h2e trainables
//   6-35: forward_mapper (emb_src, emb_dst, emb_edges, block{edge,node})
//   36-89: h_processor (emb_edges, 4 x block{edge,node})
//   90-123: backward_mapper (emb_src, emb_dst, emb_edges, block, extractor)
//   124: era_latlons 125: h_latlons 126-128: e2h/h2h/h2e attr
//   129-131: e2h/h2h/h2e edge_index (int32 [2,E])  132: z
extern "C" void kernel_launch(void* const* d_in, const int* in_sizes, int n_in,
                              void* d_out, int out_size, void* d_ws, size_t ws_size,
                              hipStream_t stream) {
  if (n_in < 133) return;
  const int ERA = 40320, Hn = 10242;
  const int Ee2h = 120960, Eh2h = 61452, Eh2e = 120960;

  auto F = [&](int i) { return (const float*)d_in[i]; };
  auto Ii = [&](int i) { return (const int*)d_in[i]; };

  char* ws = (char*)d_ws;
  size_t off = 0;
  auto alloc = [&](size_t bytes) -> void* {
    off = (off + 255) & ~(size_t)255;
    void* p = ws + off;
    off += bytes;
    return p;
  };

  auto prep_mlp = [&](int iw, int K, int M1, int M2, bool ln) -> MlpH {
    MlpH m;
    m.K = K; m.Kpad = rup(K, 32); m.M1 = M1; m.M1pad = rup(M1, 32); m.M2 = M2;
    int M1p16 = rup(M1, 16), M2p16 = rup(M2, 16);
    half_t* w0t = (half_t*)alloc((size_t)M1p16 * m.Kpad * sizeof(half_t));
    half_t* w1t = (half_t*)alloc((size_t)M2p16 * m.M1pad * sizeof(half_t));
    int n0 = M1p16 * m.Kpad;
    k_wconv<<<(n0 + THREADS - 1) / THREADS, THREADS, 0, stream>>>(F(iw), K, M1, m.Kpad, M1p16, w0t);
    int n1 = M2p16 * m.M1pad;
    k_wconv<<<(n1 + THREADS - 1) / THREADS, THREADS, 0, stream>>>(F(iw + 1), M1, M2, m.M1pad, M2p16, w1t);
    m.w0t = w0t; m.w1t = w1t;
    m.b0 = F(iw + 2); m.b1 = F(iw + 3);
    m.g = ln ? F(iw + 4) : nullptr;
    m.bn = ln ? F(iw + 5) : nullptr;
    return m;
  };

  auto run_mlp = [&](const MlpH& m,
                     const float* s0p, const int* s0i, int sw0,
                     const float* s1p, const int* s1i, int sw1,
                     const float* s2p, const int* s2i, int sw2,
                     int N, const float* resid, int rstride,
                     float* out, int final_act) {
    int grid = (N + ROWS - 1) / ROWS;
    mlp2_wmma<<<grid, THREADS, 0, stream>>>(
        s0p, s0i, sw0, s1p, s1i, sw1, s2p, s2i, sw2,
        N, m.K, m.Kpad, m.w0t, m.b0, m.M1, m.M1pad, m.w1t, m.b1, m.M2,
        m.g, m.bn, resid, rstride, out, final_act);
  };

  // ---- workspace buffers ----
  float* B_xera   = (float*)alloc((size_t)ERA * 192 * 4);   // later reused as dec dst emb [ERA,128]
  float* B_xh     = (float*)alloc((size_t)Hn * 12 * 4);
  float* B_ea_e2h = (float*)alloc((size_t)Ee2h * 11 * 4);
  float* B_ea_h2h = (float*)alloc((size_t)Eh2h * 11 * 4);
  float* B_ea_h2e = (float*)alloc((size_t)Eh2e * 11 * 4);
  float* P_xera_emb = (float*)alloc((size_t)ERA * 128 * 4); // x_era_latent; later dec node out
  float* P_xh_emb   = (float*)alloc((size_t)Hn * 128 * 4);  // later dec src emb
  float* P_xlat     = (float*)alloc((size_t)Hn * 128 * 4);
  float* R_eA  = (float*)alloc((size_t)Ee2h * 132 * 4);
  float* R_eB  = (float*)alloc((size_t)Ee2h * 132 * 4);
  float* R_x0  = (float*)alloc((size_t)Hn * 132 * 4);
  float* R_x1  = (float*)alloc((size_t)Hn * 132 * 4);
  float* R_agg = (float*)alloc((size_t)ERA * 132 * 4);

  // ---- weight prep (all f16 transposed/padded; tiny, L2 resident) ----
  MlpH enc_src = prep_mlp(6, 192, 128, 128, true);
  MlpH enc_dst = prep_mlp(12, 12, 128, 128, true);
  MlpH enc_edg = prep_mlp(18, 11, 128, 128, true);
  MlpH enc_e   = prep_mlp(24, 384, 128, 128, true);
  MlpH enc_n   = prep_mlp(30, 256, 128, 128, true);
  MlpH prc_edg = prep_mlp(36, 11, 132, 132, true);
  MlpH prc_e[4], prc_n[4];
  for (int i = 0; i < 4; ++i) {
    prc_e[i] = prep_mlp(42 + 12 * i, 396, 132, 132, true);
    prc_n[i] = prep_mlp(48 + 12 * i, 264, 132, 132, true);
  }
  MlpH dec_src = prep_mlp(90, 128, 128, 128, true);
  MlpH dec_dst = prep_mlp(96, 128, 128, 128, true);
  MlpH dec_edg = prep_mlp(102, 11, 128, 128, true);
  MlpH dec_e   = prep_mlp(108, 384, 128, 128, true);
  MlpH dec_n   = prep_mlp(114, 256, 128, 128, true);
  MlpH extr    = prep_mlp(120, 128, 128, 80, false);

  // ---- input feature builds ----
  k_build_xera<<<((ERA * 192) + THREADS - 1) / THREADS, THREADS, 0, stream>>>(
      F(0), F(124), F(1), B_xera, ERA);
  k_concat2<<<((Hn * 12) + THREADS - 1) / THREADS, THREADS, 0, stream>>>(
      F(125), 4, F(2), 8, B_xh, Hn);
  k_concat2<<<((Ee2h * 11) + THREADS - 1) / THREADS, THREADS, 0, stream>>>(
      F(126), 3, F(3), 8, B_ea_e2h, Ee2h);
  k_concat2<<<((Eh2h * 11) + THREADS - 1) / THREADS, THREADS, 0, stream>>>(
      F(127), 3, F(4), 8, B_ea_h2h, Eh2h);
  k_concat2<<<((Eh2e * 11) + THREADS - 1) / THREADS, THREADS, 0, stream>>>(
      F(128), 3, F(5), 8, B_ea_h2e, Eh2e);

  // ---- encoder (era -> h) ----
  const int* e2h = Ii(129); // row0 = era src, row1 = h dst
  run_mlp(enc_src, B_xera, nullptr, 192, nullptr, nullptr, 0, nullptr, nullptr, 0,
          ERA, nullptr, 0, P_xera_emb, 1);
  run_mlp(enc_dst, B_xh, nullptr, 12, nullptr, nullptr, 0, nullptr, nullptr, 0,
          Hn, nullptr, 0, P_xh_emb, 1);
  run_mlp(enc_edg, B_ea_e2h, nullptr, 11, nullptr, nullptr, 0, nullptr, nullptr, 0,
          Ee2h, nullptr, 0, R_eA, 1);
  run_mlp(enc_e, P_xh_emb, e2h + Ee2h, 128, P_xera_emb, e2h, 128, R_eA, nullptr, 128,
          Ee2h, R_eA, 128, R_eB, 1);
  hipMemsetAsync(R_agg, 0, (size_t)Hn * 128 * 4, stream);
  k_scatter_add<<<((Ee2h * 128) + THREADS - 1) / THREADS, THREADS, 0, stream>>>(
      R_eB, e2h + Ee2h, Ee2h, 128, R_agg);
  run_mlp(enc_n, P_xh_emb, nullptr, 128, R_agg, nullptr, 128, nullptr, nullptr, 0,
          Hn, P_xh_emb, 128, P_xlat, 1);

  // ---- processor on hidden mesh (noise-augmented, 132 channels) ----
  const int* h2h = Ii(130);
  k_concat2<<<((Hn * 132) + THREADS - 1) / THREADS, THREADS, 0, stream>>>(
      P_xlat, 128, F(132), 4, R_x0, Hn);
  run_mlp(prc_edg, B_ea_h2h, nullptr, 11, nullptr, nullptr, 0, nullptr, nullptr, 0,
          Eh2h, nullptr, 0, R_eA, 1);
  float *xc = R_x0, *xn = R_x1, *ec = R_eA, *en = R_eB;
  for (int i = 0; i < 4; ++i) {
    run_mlp(prc_e[i], xc, h2h + Eh2h, 132, xc, h2h, 132, ec, nullptr, 132,
            Eh2h, ec, 132, en, 1);
    hipMemsetAsync(R_agg, 0, (size_t)Hn * 132 * 4, stream);
    k_scatter_add<<<((Eh2h * 132) + THREADS - 1) / THREADS, THREADS, 0, stream>>>(
        en, h2h + Eh2h, Eh2h, 132, R_agg);
    run_mlp(prc_n[i], xc, nullptr, 132, R_agg, nullptr, 132, nullptr, nullptr, 0,
            Hn, xc, 132, xn, 1);
    float* t = xc; xc = xn; xn = t;
    t = ec; ec = en; en = t;
  }
  float* xcomb = xn; // free 132-wide buffer, reused as [Hn,128]
  k_combine<<<((Hn * 128) + THREADS - 1) / THREADS, THREADS, 0, stream>>>(
      xc, P_xlat, xcomb, Hn);

  // ---- decoder (h -> era) ----
  const int* h2e = Ii(131); // row0 = h src, row1 = era dst
  run_mlp(dec_src, xcomb, nullptr, 128, nullptr, nullptr, 0, nullptr, nullptr, 0,
          Hn, nullptr, 0, P_xh_emb, 1);
  float* D_dst = B_xera; // reuse [ERA,192] region as [ERA,128]
  run_mlp(dec_dst, P_xera_emb, nullptr, 128, nullptr, nullptr, 0, nullptr, nullptr, 0,
          ERA, nullptr, 0, D_dst, 1);
  run_mlp(dec_edg, B_ea_h2e, nullptr, 11, nullptr, nullptr, 0, nullptr, nullptr, 0,
          Eh2e, nullptr, 0, R_eA, 1);
  run_mlp(dec_e, D_dst, h2e + Eh2e, 128, P_xh_emb, h2e, 128, R_eA, nullptr, 128,
          Eh2e, R_eA, 128, R_eB, 1);
  hipMemsetAsync(R_agg, 0, (size_t)ERA * 128 * 4, stream);
  k_scatter_add<<<((Eh2e * 128) + THREADS - 1) / THREADS, THREADS, 0, stream>>>(
      R_eB, h2e + Eh2e, Eh2e, 128, R_agg);
  run_mlp(dec_n, D_dst, nullptr, 128, R_agg, nullptr, 128, nullptr, nullptr, 0,
          ERA, D_dst, 128, P_xera_emb, 1);

  // extractor (no final act, no LN) fused with the prognostic residual
  // x[:, :, -1, :, :80] -> resid base = x + 1*ERA*90, stride 90.
  run_mlp(extr, P_xera_emb, nullptr, 128, nullptr, nullptr, 0, nullptr, nullptr, 0,
          ERA, F(0) + (size_t)1 * ERA * 90, 90, (float*)d_out, 0);
  (void)in_sizes; (void)out_size; (void)ws_size;
}
